// lsm_network_v1_26250840113724
// MI455X (gfx1250) — matrix operand order repaired and checked by reading, not compiled, not run on hardware
//
#include <hip/hip_runtime.h>
#include <hip/hip_bf16.h>

// ---------------------------------------------------------------------------
// Problem constants (match reference)
// ---------------------------------------------------------------------------
constexpr int BATCH = 1024;   // B
constexpr int DIN   = 512;    // D_IN
constexpr int HID   = 1024;   // H
constexpr int LBL   = 32;     // L
constexpr float ALPHA_C = 0.99f;
constexpr float GAMMA_C = 1.0f / 32.0f;
constexpr int CG_ITERS_C = 30;

typedef __attribute__((ext_vector_type(16))) __bf16 v16bf;
typedef __attribute__((ext_vector_type(8)))  __bf16 v8bf;
typedef __attribute__((ext_vector_type(8)))  float  v8f;

#define DEVFN __device__ __forceinline__

DEVFN float wave_sum32(float v) {
#pragma unroll
  for (int off = 16; off > 0; off >>= 1) v += __shfl_xor(v, off, 32);
  return v;
}
DEVFN float wave_max32(float v) {
#pragma unroll
  for (int off = 16; off > 0; off >>= 1) v = fmaxf(v, __shfl_xor(v, off, 32));
  return v;
}

// ---------------------------------------------------------------------------
// fp32 -> bf16 conversion (grid-stride)
// ---------------------------------------------------------------------------
__global__ void f32_to_bf16_kernel(const float* __restrict__ src,
                                   __bf16* __restrict__ dst, size_t n) {
  size_t i = (size_t)blockIdx.x * blockDim.x + threadIdx.x;
  size_t stride = (size_t)gridDim.x * blockDim.x;
  for (; i < n; i += stride) dst[i] = (__bf16)src[i];
}

// ---------------------------------------------------------------------------
// Batched (per-label) GEMM + bias + optional ReLU using v_wmma_f32_16x16x32_bf16.
//   C[label] = act(A[label] (MxK bf16 row-major) x W[label] (KxN bf16 row-major) + bias)
//
// Block = 128 threads = 4 waves stacked in M; all 4 waves share the same n0, so
// the W (B-operand) K-tile is staged ONCE per block into LDS with
// GLOBAL_LOAD_ASYNC_TO_LDS_B128 (ASYNCcnt), double-buffered so the next tile's
// async copy overlaps the current tile's WMMAs. A fragments are per-wave-unique
// rows and stay as direct global b128 loads (+ prefetch of the next K tile).
// grid = (N/(TN*16), M/(TM*16*4), LBL)
// ---------------------------------------------------------------------------
template <int TM, int TN, bool RELU, bool OUT_BF16>
__global__ void __launch_bounds__(128)
gemm_wmma_kernel(const __bf16* __restrict__ A, size_t strideA,
                 const __bf16* __restrict__ W, size_t strideW,
                 const float* __restrict__ bias, size_t strideBias,
                 __bf16* __restrict__ outB, float* __restrict__ outF,
                 size_t strideC, int M, int N, int K) {
  constexpr int BN      = TN * 16;          // columns staged per block
  constexpr int ROWB    = BN * 2;           // bytes per LDS row (unpadded)
  constexpr int PITCH   = ROWB + 16;        // padded pitch (16B aligned)
  constexpr int CPR     = ROWB / 16;        // 16B chunks per row
  constexpr int CHUNKS  = 32 * CPR;         // chunks per K-tile
  constexpr int ITERS   = CHUNKS / 128;     // async issues per thread per tile

  __shared__ char Bsh[2][32 * PITCH];

  const int tid  = threadIdx.x;
  const int lane = tid & 31;
  const int wave = tid >> 5;
  const int label = blockIdx.z;

  A    += (size_t)label * strideA;
  W    += (size_t)label * strideW;
  bias += (size_t)label * strideBias;
  const size_t cbase = (size_t)label * strideC;

  const int m0 = (blockIdx.y * 4 + wave) * (TM * 16);
  const int n0 = blockIdx.x * BN;

  const int kb   = (lane >> 4) * 8;  // A-fragment K sub-block per lane half
  const int nl   = lane & 15;        // A row within tile / C column within tile
  const int half = lane >> 4;

  // Issue async global->LDS copy of W rows [k0, k0+32) x cols [n0, n0+BN)
  auto issue_copy = [&](int k0, int buf) {
#pragma unroll
    for (int it = 0; it < ITERS; ++it) {
      const int c   = tid + it * 128;
      const int row = c / CPR;
      const int cir = c % CPR;
      const unsigned lds_off =
          (unsigned)(size_t)(&Bsh[buf][0]) + (unsigned)(row * PITCH + cir * 16);
      const __bf16* gp = W + (size_t)(k0 + row) * (size_t)N + (size_t)(n0 + cir * 8);
      asm volatile("global_load_async_to_lds_b128 %0, %1, off"
                   :: "v"(lds_off), "v"(gp)
                   : "memory");
    }
  };

  v8f c[TM][TN];
#pragma unroll
  for (int i = 0; i < TM; ++i)
#pragma unroll
    for (int j = 0; j < TN; ++j)
#pragma unroll
      for (int r = 0; r < 8; ++r) c[i][j][r] = 0.0f;

  issue_copy(0, 0);

  const int ktiles = K / 32;
  for (int t = 0; t < ktiles; ++t) {
    const int k0 = t * 32;

    // Overlap: kick off next tile's copy, then drain the previous tile only.
    if (t + 1 < ktiles) {
      issue_copy(k0 + 32, (t + 1) & 1);
      asm volatile("s_wait_asynccnt %0" :: "n"(ITERS) : "memory");
    } else {
      asm volatile("s_wait_asynccnt 0" ::: "memory");
    }
    __syncthreads();  // tile t visible to all 4 waves

    // A fragments (global): lane holds A[m0+i*16+nl, k0+kb+{0..7 , 16..23}]
    v16bf afrag[TM];
#pragma unroll
    for (int i = 0; i < TM; ++i) {
      const __bf16* ap = A + (size_t)(m0 + i * 16 + nl) * (size_t)K + (size_t)(k0 + kb);
      v8bf lo = *(const v8bf*)ap;
      v8bf hi = *(const v8bf*)(ap + 16);
      afrag[i] = __builtin_shufflevector(lo, hi, 0, 1, 2, 3, 4, 5, 6, 7,
                                         8, 9, 10, 11, 12, 13, 14, 15);
      __builtin_prefetch(ap + 64, 0, 1);  // next K tile (speculative)
    }

    // B fragments (LDS): lane = N column; K rows kb+{0..7, 16..23}.
    const char* bcol = &Bsh[t & 1][0] + nl * 2;
    v16bf bfrag[TN];
#pragma unroll
    for (int j = 0; j < TN; ++j) {
#pragma unroll
      for (int s = 0; s < 16; ++s) {
        const int krow = kb + (s < 8 ? s : s + 8);
        bfrag[j][s] = *(const __bf16*)(bcol + krow * PITCH + j * 32);
      }
    }

#pragma unroll
    for (int i = 0; i < TM; ++i)
#pragma unroll
      for (int j = 0; j < TN; ++j)
        c[i][j] = __builtin_amdgcn_wmma_f32_16x16x32_bf16(
            false, afrag[i], false, bfrag[j], (short)0, c[i][j], false, false);

    __syncthreads();  // all waves done reading buf[t&1] before it is reused
  }

  // Epilogue: bias + activation + store. C layout: VGPR r -> M = r + 8*half, N = nl
#pragma unroll
  for (int i = 0; i < TM; ++i) {
#pragma unroll
    for (int j = 0; j < TN; ++j) {
      const float bv = bias[n0 + j * 16 + nl];
#pragma unroll
      for (int r = 0; r < 8; ++r) {
        float v = c[i][j][r] + bv;
        if (RELU) v = fmaxf(v, 0.0f);
        const size_t m = (size_t)(m0 + i * 16 + r + 8 * half);
        const size_t idx = cbase + m * (size_t)N + (size_t)(n0 + j * 16 + nl);
        if (OUT_BF16) outB[idx] = (__bf16)v;
        else          outF[idx] = v;
      }
    }
  }
}

// ---------------------------------------------------------------------------
// Sigmoid heads: y1[b,l] = sigmoid(dot(Z[l,b,:], Wo[l,:,0]) + bo[l,0])
// Z layout: [LBL, BATCH, 32] fp32. Writes y1 (out sec 2) and y_hat (out sec 0).
// ---------------------------------------------------------------------------
__global__ void head_kernel(const float* __restrict__ Z, const float* __restrict__ Wo,
                            const float* __restrict__ bo, float* __restrict__ out) {
  const int idx = blockIdx.x * blockDim.x + threadIdx.x;  // = b*32 + l
  const int b = idx >> 5;
  const int l = idx & 31;
  const float* z = Z + ((size_t)l * BATCH + b) * 32;
  float s = bo[l * 2 + 0];
#pragma unroll
  for (int k = 0; k < 32; ++k) s += z[k] * Wo[(l * 32 + k) * 2 + 0];
  const float y1 = 1.0f / (1.0f + expf(-s));
  out[2 * BATCH * LBL + idx] = y1;                      // y1
  out[idx] = (y1 >= 0.5f) ? 1.0f : 0.0f;                // y_hat
}

// ---------------------------------------------------------------------------
// Affinity + normalized Laplacian: builds Amat[b] (32x32 fp32) per batch.
// One wave per batch, 8 waves per block.
// ---------------------------------------------------------------------------
__global__ void __launch_bounds__(256)
affinity_kernel(const float* __restrict__ Z, const float* __restrict__ mu,
                float* __restrict__ Amat) {
  __shared__ float Ksh[8][32][33];
  const int lane = threadIdx.x & 31;
  const int wave = threadIdx.x >> 5;
  const int b = blockIdx.x * 8 + wave;

  // zn[b, lane] = sum_k Z[lane, b, k];  mn[b, lane] = sum_k mu[b, lane, k]
  float zn = 0.0f;
  {
    const float* zp = Z + ((size_t)lane * BATCH + b) * 32;
#pragma unroll
    for (int k = 0; k < 32; ++k) zn += zp[k];
  }
  float mn = 0.0f;
  {
    const float* mp = mu + ((size_t)b * 32 + lane) * 32;
#pragma unroll
    for (int k = 0; k < 32; ++k) mn += mp[k];
  }

  // K row (lane = i): K[i][j] = exp(-gamma*(zn_i + mn_j - 2 zn_i mn_j))
#pragma unroll
  for (int j = 0; j < 32; ++j) {
    const float mnj = __shfl(mn, j, 32);
    const float t = zn + mnj - 2.0f * zn * mnj;
    Ksh[wave][lane][j] = expf(-GAMMA_C * t);
  }
  __syncthreads();

  // S[j] = sum_i K[i][j]  (lane = j)
  float S = 0.0f;
#pragma unroll
  for (int i = 0; i < 32; ++i) S += Ksh[wave][i][lane];
  const float w = rsqrtf(S);

  // A[i][j] = I - alpha * w_i K_ij w_j + 1e-6 I   (lane = i)
  const float wi = w;
  float* arow = Amat + ((size_t)b * 32 + lane) * 32;
#pragma unroll
  for (int j = 0; j < 32; ++j) {
    const float wj = __shfl(w, j, 32);
    float v = -ALPHA_C * wi * Ksh[wave][lane][j] * wj;
    if (j == lane) v += 1.0f + 1e-6f;
    arow[j] = v;
  }
}

// ---------------------------------------------------------------------------
// Batched CG: one wave per batch, A row in registers (lane = row).
// rhs = y_hat (out sec 0); writes yhat (out sec 1).
// ---------------------------------------------------------------------------
__global__ void __launch_bounds__(256)
cg_kernel(const float* __restrict__ Amat, float* __restrict__ out) {
  const int lane = threadIdx.x & 31;
  const int wave = threadIdx.x >> 5;
  const int b = blockIdx.x * 8 + wave;

  float Arow[32];
  {
    const float* ap = Amat + ((size_t)b * 32 + lane) * 32;
#pragma unroll
    for (int j = 0; j < 32; ++j) Arow[j] = ap[j];
  }
  const float rhs = out[b * 32 + lane];  // y_hat
  float x = 0.0f, r = rhs, p = rhs;
  float rs = wave_sum32(r * r);

  for (int it = 0; it < CG_ITERS_C; ++it) {
    float Ap = 0.0f;
#pragma unroll
    for (int j = 0; j < 32; ++j) Ap += Arow[j] * __shfl(p, j, 32);
    const float pAp = wave_sum32(p * Ap);
    const float a = rs / (pAp + 1e-30f);
    x += a * p;
    r -= a * Ap;
    const float rs_new = wave_sum32(r * r);
    const float beta = rs_new / (rs + 1e-30f);
    p = r + beta * p;
    rs = rs_new;
  }
  out[BATCH * LBL + b * 32 + lane] = (x >= 0.5f) ? 1.0f : 0.0f;  // yhat
}

// ---------------------------------------------------------------------------
// Attention pooling + final softmax. One wave per batch.
// Writes Y_pred (out sec 3).
// ---------------------------------------------------------------------------
__global__ void __launch_bounds__(256)
attention_kernel(const float* __restrict__ Z, const float* __restrict__ Wk,
                 const float* __restrict__ bk, const float* __restrict__ Wd,
                 const float* __restrict__ bd, const float* __restrict__ Wf,
                 const float* __restrict__ bf, float* __restrict__ out) {
  const int lane = threadIdx.x & 31;
  const int wave = threadIdx.x >> 5;
  const int b = blockIdx.x * 8 + wave;

  // Phase 1 (lane = l): att_s[b,l] = sum_k tanh(dot(Z[b,l,:], Wk[:,k]) + bk[k]) * Wd[k] + bd
  float zv[32];
  {
    const float* zp = Z + ((size_t)lane * BATCH + b) * 32;
#pragma unroll
    for (int h = 0; h < 32; ++h) zv[h] = zp[h];
  }
  float s = 0.0f;
#pragma unroll 4
  for (int k = 0; k < 32; ++k) {
    float t = bk[k];
#pragma unroll
    for (int h = 0; h < 32; ++h) t += zv[h] * Wk[h * 32 + k];
    s += tanhf(t) * Wd[k];
  }
  const float att_s = s + bd[0];

  // softmax over label lanes
  const float mx = wave_max32(att_s);
  const float e = expf(att_s - mx);
  const float denom = wave_sum32(e);
  const float aw = e / denom;

  // Phase 2 (lane = h): ao[h] = sum_l Z[b,l,h] * aw[l]
  float ao = 0.0f;
#pragma unroll
  for (int l = 0; l < 32; ++l) {
    const float awl = __shfl(aw, l, 32);
    ao += Z[((size_t)l * BATCH + b) * 32 + lane] * awl;
  }

  // Phase 3 (lane = j): logits[j] = sum_h ao[h]*Wf[h,j] + bf[j]; softmax
  float logit = bf[lane];
#pragma unroll
  for (int h = 0; h < 32; ++h) logit += __shfl(ao, h, 32) * Wf[h * 32 + lane];
  const float lm = wave_max32(logit);
  const float le = expf(logit - lm);
  const float ls = wave_sum32(le);
  out[3 * BATCH * LBL + b * 32 + lane] = le / ls;  // Y_pred
}

// ---------------------------------------------------------------------------
// Host launcher
// ---------------------------------------------------------------------------
extern "C" void kernel_launch(void* const* d_in, const int* in_sizes, int n_in,
                              void* d_out, int out_size, void* d_ws, size_t ws_size,
                              hipStream_t stream) {
  const float* x  = (const float*)d_in[0];
  const float* mu = (const float*)d_in[1];
  const float* W0 = (const float*)d_in[2];
  const float* b0 = (const float*)d_in[3];
  const float* W1 = (const float*)d_in[4];
  const float* b1 = (const float*)d_in[5];
  const float* W2 = (const float*)d_in[6];
  const float* b2 = (const float*)d_in[7];
  const float* Wo = (const float*)d_in[8];
  const float* bo = (const float*)d_in[9];
  const float* Wk = (const float*)d_in[10];
  const float* bk = (const float*)d_in[11];
  const float* Wd = (const float*)d_in[12];
  const float* bd = (const float*)d_in[13];
  const float* Wf = (const float*)d_in[14];
  const float* bf = (const float*)d_in[15];
  float* out = (float*)d_out;

  // Workspace carving (256B aligned)
  char* ws = (char*)d_ws;
  size_t off = 0;
  auto carve = [&](size_t bytes) -> char* {
    char* p = ws + off;
    off = (off + bytes + 255) & ~(size_t)255;
    return p;
  };
  constexpr size_t N_XBF  = (size_t)BATCH * DIN;        // x bf16
  constexpr size_t N_W0   = (size_t)LBL * DIN * HID;    // W0 bf16
  constexpr size_t N_W1   = (size_t)LBL * HID * HID;    // W1 bf16
  constexpr size_t N_W2   = (size_t)LBL * HID * LBL;    // W2 bf16
  constexpr size_t N_H    = (size_t)LBL * BATCH * HID;  // h0/h1 bf16 ([L,B,H])
  constexpr size_t N_Z    = (size_t)LBL * BATCH * LBL;  // Z fp32 ([L,B,32])
  constexpr size_t N_AMAT = (size_t)BATCH * LBL * LBL;  // Amat fp32

  __bf16* x_bf  = (__bf16*)carve(N_XBF * 2);
  __bf16* W0_bf = (__bf16*)carve(N_W0 * 2);
  __bf16* W1_bf = (__bf16*)carve(N_W1 * 2);
  __bf16* W2_bf = (__bf16*)carve(N_W2 * 2);
  __bf16* h0_bf = (__bf16*)carve(N_H * 2);
  __bf16* h1_bf = (__bf16*)carve(N_H * 2);
  float*  Z     = (float*)carve(N_Z * 4);
  float*  Amat  = (float*)carve(N_AMAT * 4);

  // 1) fp32 -> bf16 conversions
  f32_to_bf16_kernel<<<1024, 256, 0, stream>>>(x,  x_bf,  N_XBF);
  f32_to_bf16_kernel<<<4096, 256, 0, stream>>>(W0, W0_bf, N_W0);
  f32_to_bf16_kernel<<<8192, 256, 0, stream>>>(W1, W1_bf, N_W1);
  f32_to_bf16_kernel<<<1024, 256, 0, stream>>>(W2, W2_bf, N_W2);

  // 2) Stage 1: h0[l] = relu(x @ W0[l] + b0[l])   M=1024 N=1024 K=512
  {
    dim3 grid(HID / 64, BATCH / 256, LBL);
    gemm_wmma_kernel<4, 4, true, true><<<grid, 128, 0, stream>>>(
        x_bf, (size_t)0, W0_bf, (size_t)DIN * HID, b0, (size_t)HID,
        h0_bf, nullptr, (size_t)BATCH * HID, BATCH, HID, DIN);
  }
  // 3) Stage 2: h1[l] = relu(h0[l] @ W1[l] + b1[l])   M=1024 N=1024 K=1024
  {
    dim3 grid(HID / 64, BATCH / 256, LBL);
    gemm_wmma_kernel<4, 4, true, true><<<grid, 128, 0, stream>>>(
        h0_bf, (size_t)BATCH * HID, W1_bf, (size_t)HID * HID, b1, (size_t)HID,
        h1_bf, nullptr, (size_t)BATCH * HID, BATCH, HID, HID);
  }
  // 4) Stage 3: Z[l] = relu(h1[l] @ W2[l] + b2[l])   M=1024 N=32 K=1024 (fp32 out)
  {
    dim3 grid(1, BATCH / 256, LBL);
    gemm_wmma_kernel<4, 2, true, false><<<grid, 128, 0, stream>>>(
        h1_bf, (size_t)BATCH * HID, W2_bf, (size_t)HID * LBL, b2, (size_t)LBL,
        nullptr, Z, (size_t)BATCH * LBL, BATCH, LBL, HID);
  }

  // 5) Heads -> y_hat (sec 0), y1 (sec 2)
  head_kernel<<<(BATCH * LBL) / 256, 256, 0, stream>>>(Z, Wo, bo, out);

  // 6) Affinity -> Amat
  affinity_kernel<<<BATCH / 8, 256, 0, stream>>>(Z, mu, Amat);

  // 7) CG -> yhat (sec 1)
  cg_kernel<<<BATCH / 8, 256, 0, stream>>>(Amat, out);

  // 8) Attention -> Y_pred (sec 3)
  attention_kernel<<<BATCH / 8, 256, 0, stream>>>(Z, Wk, bk, Wd, bd, Wf, bf, out);

  (void)in_sizes; (void)n_in; (void)out_size; (void)ws_size;
}
